// DualStateHSGBDH_29171417874543
// MI455X (gfx1250) — matrix operand order repaired and checked by reading, not compiled, not run on hardware
//
#include <hip/hip_runtime.h>
#include <stdint.h>
#include <stddef.h>

// ---------------------------------------------------------------------------
// DualStateHSGBDH on MI455X (gfx1250, wave32)
//   B=8, T=48, D=512, N=1024
// ---------------------------------------------------------------------------

typedef __attribute__((ext_vector_type(4)))  float   f4;
typedef __attribute__((ext_vector_type(8)))  float   v8f;
typedef __attribute__((ext_vector_type(16))) __bf16  v16bf;

#define B_  8
#define T_  48
#define D_  512
#define N_  1024
#define ROWS_ (B_ * T_ * 2)          // 768 packed rows (xn part + tn part)

static __device__ __forceinline__ unsigned short f2bf(float f) {
  union { float f; unsigned u; } v; v.f = f;
  unsigned r = v.u + 0x7FFFu + ((v.u >> 16) & 1u);   // round-to-nearest-even
  return (unsigned short)(r >> 16);
}

// ---- prep: pack x_seq rows and shifted-target rows into one bf16 matrix ----
__global__ void build_u_kernel(const float* __restrict__ x_seq,
                               const float* __restrict__ targets,
                               unsigned short* __restrict__ Ubf) {
  int idx = blockIdx.x * 256 + threadIdx.x;          // over 768*512
  if (idx >= ROWS_ * D_) return;
  int r = idx >> 9;                                  // /512
  int k = idx & 511;
  float v;
  if (r < B_ * T_) {
    v = x_seq[idx];                                  // row r == b*T+t, col k
  } else {
    int r2 = r - B_ * T_;
    int t  = r2 % T_;
    v = (t < T_ - 1) ? targets[((r2 + 1) << 9) + k] : 0.0f;  // targets[b,t+1,:]
  }
  Ubf[idx] = f2bf(v);
}

// ---- prep: transpose+convert a [rows,cols] f32 matrix to bf16 [cols,rows] --
__global__ void transp_bf_kernel(const float* __restrict__ src,
                                 unsigned short* __restrict__ dst,
                                 int rows, int cols) {
  int idx = blockIdx.x * 256 + threadIdx.x;
  if (idx >= rows * cols) return;
  int r = idx / cols;
  int c = idx - r * cols;
  dst[(size_t)c * rows + r] = f2bf(src[idx]);
}

// ---- prep: Gp = 0.5*I per batch, Gt = 0 (float4-wide writes) ---------------
__global__ void init_state_kernel(f4* __restrict__ Gp4, f4* __restrict__ Gt4) {
  int i = blockIdx.x * 256 + threadIdx.x;            // over N*N*B/4 quarters
  const int total4 = (B_ * N_ * N_) / 4;             // 2097152
  if (i >= total4) return;
  size_t f0 = (size_t)i * 4;
  int m0 = (int)(f0 & (N_ - 1));
  int n  = (int)((f0 >> 10) & (N_ - 1));
  f4 g = {0.f, 0.f, 0.f, 0.f};
  if (n >= m0 && n < m0 + 4) ((float*)&g)[n - m0] = 0.5f;
  f4 z = {0.f, 0.f, 0.f, 0.f};
  Gp4[i] = g;
  Gt4[i] = z;
}

__global__ void zero_kernel(float* __restrict__ p, int n) {
  int i = blockIdx.x * 256 + threadIdx.x;
  if (i < n) p[i] = 0.0f;
}

__global__ void r2bf_kernel(const float* __restrict__ R,
                            unsigned short* __restrict__ Rbf, int n) {
  int i = blockIdx.x * 256 + threadIdx.x;
  if (i < n) Rbf[i] = f2bf(R[i]);
}

// ---------------------------------------------------------------------------
// WMMA GEMM:  C[M,Nn] (f32, optional relu) = A[M,K](bf16) @ Bt[Nn,K](bf16)^T
// Block = 256 threads = 8 waves; each wave owns one 16x16 tile.
// Block tile: 16 rows x 128 cols.  Grid: (M/16, Nn/128).
// Fragment layouts per CDNA5 ISA 7.12.2 (16-bit A 16x32 / B 32x16, wave32).
// ---------------------------------------------------------------------------
__global__ void wmma_gemm_kernel(const unsigned short* __restrict__ A,
                                 const unsigned short* __restrict__ Bt,
                                 float* __restrict__ C,
                                 int M, int Nn, int K, int do_relu) {
  const int lane = threadIdx.x & 31;
  const int wave = threadIdx.x >> 5;
  const int row0 = blockIdx.x * 16;
  const int col0 = blockIdx.y * 128 + wave * 16;
  const int mr = lane & 15;          // A: M index ; B: N index ; D: col index
  const int h  = lane >> 4;          // lane half selects K sub-range

  union Frag { f4 q[2]; v16bf v; };

  const f4* Arow = (const f4*)(A + (size_t)(row0 + mr) * K);
  const f4* Brow = (const f4*)(Bt + (size_t)(col0 + mr) * K);

  v8f c = {0.f, 0.f, 0.f, 0.f, 0.f, 0.f, 0.f, 0.f};

  for (int kb = 0; kb < K; kb += 32) {
    int base4 = kb >> 3;             // float4 index of halfword offset kb
    Frag a, b;
    // A lane holds K = kb+8h..+7 (elems 0..7) and kb+16+8h..+7 (elems 8..15)
    a.q[0] = Arow[base4 + h];
    a.q[1] = Arow[base4 + 2 + h];
    // B lane holds K = kb+8h..+7 (elems 0..7) and kb+16+8h..+7 (elems 8..15)
    b.q[0] = Brow[base4 + h];
    b.q[1] = Brow[base4 + 2 + h];
    c = __builtin_amdgcn_wmma_f32_16x16x32_bf16(false, a.v, false, b.v,
                                                (short)0, c, false, false);
  }

  // D layout: VGPR r -> row = r + 8*h ; col = lane&15
  const int cn = col0 + mr;
  #pragma unroll
  for (int r = 0; r < 8; ++r) {
    int rowi = row0 + r + 8 * h;
    float val = c[r];
    if (do_relu) val = fmaxf(val, 0.0f);
    C[(size_t)rowi * Nn + cn] = val;
  }
}

// ---- row-wise layernorm + relu over P[768,1024] -> XN[384,1024], TN[...] ---
__global__ void ln_relu_kernel(const float* __restrict__ P,
                               float* __restrict__ XN,
                               float* __restrict__ TN) {
  const int row = blockIdx.x;                        // 0..767
  const int tid = threadIdx.x;                       // 256
  const float* p = P + (size_t)row * N_;
  float v[4], s = 0.f, s2 = 0.f;
  #pragma unroll
  for (int j = 0; j < 4; ++j) {
    v[j] = p[tid + 256 * j];
    s  += v[j];
    s2 += v[j] * v[j];
  }
  __shared__ float red[256], red2[256];
  red[tid] = s; red2[tid] = s2;
  __syncthreads();
  for (int off = 128; off > 0; off >>= 1) {
    if (tid < off) { red[tid] += red[tid + off]; red2[tid] += red2[tid + off]; }
    __syncthreads();
  }
  float mu  = red[0] * (1.0f / N_);
  float var = red2[0] * (1.0f / N_) - mu * mu;
  float rstd = rsqrtf(var + 1e-5f);
  float* out = (row < B_ * T_) ? (XN + (size_t)row * N_)
                               : (TN + (size_t)(row - B_ * T_) * N_);
  #pragma unroll
  for (int j = 0; j < 4; ++j)
    out[tid + 256 * j] = fmaxf((v[j] - mu) * rstd, 0.0f);
}

// ---------------------------------------------------------------------------
// One scan step, fused: for each (b, n-chunk) block sweep 32 rows of Gp/Gt,
// accumulating r_p/r_t partials for all 1024 columns and applying the
// max-outer-product update in the same pass (read old -> use -> write new).
// Partial r merged into R[b,t,:] with global_atomic_add_f32, alpha folded in.
// ---------------------------------------------------------------------------
__global__ void step_kernel(const float* __restrict__ XN,
                            const float* __restrict__ TN,
                            float* __restrict__ Gp,
                            float* __restrict__ Gt,
                            float* __restrict__ R,
                            const float* __restrict__ mixp,
                            int t, int upd) {
  const int b     = blockIdx.y;
  const int chunk = blockIdx.x;                      // 0..31, 32 rows each
  const int tid   = threadIdx.x;                     // 256 -> 4 cols/thread

  __shared__ float s_xn[N_];
  __shared__ float s_tn[N_];
  const float* xn = XN + ((size_t)b * T_ + t) * N_;
  const float* tn = TN + ((size_t)b * T_ + t) * N_;
  for (int j = tid; j < N_; j += 256) { s_xn[j] = xn[j]; s_tn[j] = tn[j]; }
  __syncthreads();

  const int m0 = tid * 4;
  const f4 xm = *(const f4*)&s_xn[m0];
  const f4 tm = *(const f4*)&s_tn[m0];

  f4* Gp4 = (f4*)(Gp + (size_t)b * N_ * N_);
  f4* Gt4 = (f4*)(Gt + (size_t)b * N_ * N_);

  f4 rp = {0.f, 0.f, 0.f, 0.f};
  f4 rt = {0.f, 0.f, 0.f, 0.f};

  const int n0 = chunk * 32;
  #pragma unroll 4
  for (int i = 0; i < 32; ++i) {
    const int n = n0 + i;
    const float xnn = s_xn[n];
    const size_t idx = (size_t)n * (N_ / 4) + tid;
    f4 gp = Gp4[idx];
    f4 gt = Gt4[idx];
    rp.x += xnn * gp.x; rp.y += xnn * gp.y; rp.z += xnn * gp.z; rp.w += xnn * gp.w;
    rt.x += xnn * gt.x; rt.y += xnn * gt.y; rt.z += xnn * gt.z; rt.w += xnn * gt.w;
    if (upd) {
      const float hx = 0.5f * xnn;
      gp.x = fmaxf(gp.x, hx * xm.x); gp.y = fmaxf(gp.y, hx * xm.y);
      gp.z = fmaxf(gp.z, hx * xm.z); gp.w = fmaxf(gp.w, hx * xm.w);
      gt.x = fmaxf(gt.x, hx * tm.x); gt.y = fmaxf(gt.y, hx * tm.y);
      gt.z = fmaxf(gt.z, hx * tm.z); gt.w = fmaxf(gt.w, hx * tm.w);
      Gp4[idx] = gp;
      Gt4[idx] = gt;
    }
  }

  const float mp = mixp[0];
  const float alpha = 1.0f / (1.0f + __expf(-mp));
  const float beta = 1.0f - alpha;
  float* r = R + ((size_t)b * T_ + t) * N_ + m0;
  unsafeAtomicAdd(&r[0], alpha * rp.x + beta * rt.x);
  unsafeAtomicAdd(&r[1], alpha * rp.y + beta * rt.y);
  unsafeAtomicAdd(&r[2], alpha * rp.z + beta * rt.z);
  unsafeAtomicAdd(&r[3], alpha * rp.w + beta * rt.w);
}

// ---------------------------------------------------------------------------
extern "C" void kernel_launch(void* const* d_in, const int* in_sizes, int n_in,
                              void* d_out, int out_size, void* d_ws, size_t ws_size,
                              hipStream_t stream) {
  const float* x_seq   = (const float*)d_in[0];
  const float* targets = (const float*)d_in[1];
  const float* E       = (const float*)d_in[2];
  const float* Dy      = (const float*)d_in[3];
  const float* mixp    = (const float*)d_in[4];
  float* out = (float*)d_out;

  char* ws = (char*)d_ws;
  // workspace layout (bytes, 16B aligned)
  float*          Gp    = (float*)(ws + 0);                       // 32 MB
  float*          Gt    = (float*)(ws + 33554432);                // 32 MB
  float*          XN    = (float*)(ws + 67108864);                // 1.5 MB
  float*          TN    = (float*)(ws + 68681728);                // 1.5 MB
  float*          R     = (float*)(ws + 70254592);                // 1.5 MB
  float*          P     = (float*)(ws + 71827456);                // 3 MB
  unsigned short* Ubf   = (unsigned short*)(ws + 74973184);       // 0.75 MB
  unsigned short* EbfT  = (unsigned short*)(ws + 75759616);       // 1 MB
  unsigned short* DybfT = (unsigned short*)(ws + 76808192);       // 1 MB
  unsigned short* Rbf   = (unsigned short*)(ws + 77856768);       // 0.75 MB

  // --- prep ---
  build_u_kernel<<<(ROWS_ * D_ + 255) / 256, 256, 0, stream>>>(x_seq, targets, Ubf);
  transp_bf_kernel<<<(D_ * N_ + 255) / 256, 256, 0, stream>>>(E,  EbfT,  D_, N_);
  transp_bf_kernel<<<(N_ * D_ + 255) / 256, 256, 0, stream>>>(Dy, DybfT, N_, D_);
  init_state_kernel<<<(B_ * N_ * N_ / 4 + 255) / 256, 256, 0, stream>>>((f4*)Gp, (f4*)Gt);
  zero_kernel<<<(B_ * T_ * N_ + 255) / 256, 256, 0, stream>>>(R, B_ * T_ * N_);

  // --- P[768,1024] = Ubf @ E  (WMMA bf16) ---
  {
    dim3 grid(ROWS_ / 16, N_ / 128);
    wmma_gemm_kernel<<<grid, 256, 0, stream>>>(Ubf, EbfT, P, ROWS_, N_, D_, 0);
  }
  // --- layernorm + relu -> XN, TN ---
  ln_relu_kernel<<<ROWS_, 256, 0, stream>>>(P, XN, TN);

  // --- sequential scan over T, fused GEMV + max-update per step ---
  for (int t = 0; t < T_; ++t) {
    dim3 grid(32, B_);
    step_kernel<<<grid, 256, 0, stream>>>(XN, TN, Gp, Gt, R, mixp, t,
                                          (t < T_ - 1) ? 1 : 0);
  }

  // --- Y = relu(R @ Dy) -> d_out  (WMMA bf16) ---
  r2bf_kernel<<<(B_ * T_ * N_ + 255) / 256, 256, 0, stream>>>(R, Rbf, B_ * T_ * N_);
  {
    dim3 grid((B_ * T_) / 16, D_ / 128);
    wmma_gemm_kernel<<<grid, 256, 0, stream>>>(Rbf, DybfT, out, B_ * T_, D_, N_, 1);
  }
  (void)in_sizes; (void)n_in; (void)out_size; (void)ws_size;
}
